// L4maAttention_3539053052074
// MI455X (gfx1250) — matrix operand order repaired and checked by reading, not compiled, not run on hardware
//
#include <hip/hip_runtime.h>
#include <hip/hip_bf16.h>
#include <stdint.h>

// ---------------- problem constants (fixed shapes from the reference) ------
#define BB    4
#define QO    512
#define PAGE  16
#define PPS   128
#define HID   4096
#define HQ    32
#define HKV   8
#define DH    128
#define NTOK  (BB*QO)          // 2048
#define SEQ   (PPS*PAGE)       // 2048
#define HIST  (SEQ-QO)         // 1536 cached positions per sequence

typedef __attribute__((ext_vector_type(16))) __bf16 v16bf;
typedef __attribute__((ext_vector_type(8)))  float  v8f;

// ---------------------------------------------------------------------------
// CDNA5 async global->LDS copy (ASYNCcnt-tracked). GV addressing mode:
//   VDST = VGPR with LDS byte offset, VADDR = 64-bit global address, SADDR=off
// ---------------------------------------------------------------------------
__device__ __forceinline__ void async_load_lds_b128(unsigned lds_off,
                                                    const void* gaddr) {
  asm volatile("global_load_async_to_lds_b128 %0, %1, off"
               :
               : "v"(lds_off), "v"((unsigned long long)(uintptr_t)gaddr)
               : "memory");
}
__device__ __forceinline__ void wait_asynccnt0() {
  asm volatile("s_wait_asynccnt 0x0" ::: "memory");
}

// ---------------------------------------------------------------------------
// f32 -> bf16 elementwise convert
// ---------------------------------------------------------------------------
__global__ __launch_bounds__(256) void cvt_f32_bf16_kernel(
    const float* __restrict__ x, __bf16* __restrict__ y, int n) {
  int i = blockIdx.x * blockDim.x + threadIdx.x;
  if (i < n) y[i] = (__bf16)x[i];
}

// ---------------------------------------------------------------------------
// GEMM: C[M,Nout] = A[M,K](bf16, row major) * W[Nout,K](f32, row major)^T
// 256 threads = 8 waves; each wave -> one 16(M) x 64(N) tile via 4 WMMAs/kstep.
// W is converted fp32->bf16 in registers (weights are read ~once; L2 192MB
// absorbs cross-M-tile re-reads).
// ---------------------------------------------------------------------------
__global__ __launch_bounds__(256) void gemm_bf16_kernel(
    const __bf16* __restrict__ A, const float* __restrict__ W,
    float* __restrict__ C, int M, int K, int Nout) {
  const int wave = threadIdx.x >> 5;
  const int lane = threadIdx.x & 31;
  const int half = lane >> 4;          // 0: lanes 0-15, 1: lanes 16-31
  const int lm   = lane & 15;
  const int m0 = blockIdx.x * 16;
  const int n0 = blockIdx.y * 512 + wave * 64;
  if (m0 >= M || n0 >= Nout) return;

  v8f acc[4] = {};
  const __bf16* arow = A + (size_t)(m0 + lm) * K + half * 8;

  for (int k0 = 0; k0 < K; k0 += 32) {
    // A fragment (16x32): lane holds row m0+lm; element e -> K = k0 +
    //   (e<8 ? e : e+8) + half*8   (ISA 16-bit A layout)
    v16bf a;
    {
      const __bf16* p = arow + k0;
#pragma unroll
      for (int e = 0; e < 8; ++e) { a[e] = p[e]; a[e + 8] = p[e + 16]; }
    }
#pragma unroll
    for (int t = 0; t < 4; ++t) {
      // B fragment (32x16): lane -> col n0+t*16+lm ; element e -> K =
      //   k0 + (e<8 ? e : e+8) + half*8 (fp32, convert to bf16 in regs)
      const float* wr = W + (size_t)(n0 + t * 16 + lm) * K + k0;
      if (t == 0) __builtin_prefetch((const void*)(wr + 32), 0, 1);
      v16bf b;
#pragma unroll
      for (int e = 0; e < 8; ++e) { b[e] = (__bf16)wr[e + half * 8]; b[e + 8] = (__bf16)wr[e + 16 + half * 8]; }
      acc[t] = __builtin_amdgcn_wmma_f32_16x16x32_bf16(
          false, a, false, b, (short)0, acc[t], false, false);
    }
  }
  // C layout: VGPR j -> row m0 + half*8 + j ; lane lm -> col
#pragma unroll
  for (int t = 0; t < 4; ++t) {
    float* c0 = C + (size_t)(m0 + half * 8) * Nout + n0 + t * 16 + lm;
#pragma unroll
    for (int j = 0; j < 8; ++j) c0[(size_t)j * Nout] = acc[t][j];
  }
}

// ---------------------------------------------------------------------------
// llama3.1 scaled-rope inverse frequency for pair index i (0..63)
// ---------------------------------------------------------------------------
__device__ __forceinline__ float llama31_freq(int i) {
  float inv = __powf(5.0e5f, -(2.0f * (float)i) / 128.0f);
  float wavelen = 6.2831853071795864f / inv;
  float smooth = (8192.0f / wavelen - 1.0f) * (1.0f / 3.0f);
  float mid = (1.0f - smooth) * inv * 0.125f + smooth * inv;
  return (wavelen > 8192.0f) ? inv * 0.125f
                             : ((wavelen < 2048.0f) ? inv : mid);
}

// RoPE q (f32 [n,HQ,DH]) -> bf16 [b][h][t][DH]
__global__ __launch_bounds__(256) void rope_q_kernel(
    const float* __restrict__ q, const int* __restrict__ pos,
    __bf16* __restrict__ qb) {
  int idx = blockIdx.x * blockDim.x + threadIdx.x;   // N*HQ*64
  if (idx >= NTOK * HQ * 64) return;
  int i = idx & 63;
  int h = (idx >> 6) & (HQ - 1);
  int n = idx >> 11;
  float f = llama31_freq(i);
  float s, c;
  __sincosf((float)pos[n] * f, &s, &c);
  const float* x = q + ((size_t)n * HQ + h) * DH;
  float x1 = x[i], x2 = x[i + 64];
  int b = n / QO, t = n % QO;
  __bf16* dst = qb + (((size_t)b * HQ + h) * QO + t) * DH;
  dst[i]      = (__bf16)(x1 * c - x2 * s);
  dst[i + 64] = (__bf16)(x2 * c + x1 * s);
}

// RoPE k (f32 [n,HKV,DH]) -> bf16 K [b][h][1536+t][DH]
__global__ __launch_bounds__(256) void rope_k_kernel(
    const float* __restrict__ k, const int* __restrict__ pos,
    __bf16* __restrict__ kb) {
  int idx = blockIdx.x * blockDim.x + threadIdx.x;   // N*HKV*64
  if (idx >= NTOK * HKV * 64) return;
  int i = idx & 63;
  int h = (idx >> 6) & (HKV - 1);
  int n = idx >> 9;
  float f = llama31_freq(i);
  float s, c;
  __sincosf((float)pos[n] * f, &s, &c);
  const float* x = k + ((size_t)n * HKV + h) * DH;
  float x1 = x[i], x2 = x[i + 64];
  int b = n / QO, t = n % QO;
  __bf16* dst = kb + (((size_t)b * HKV + h) * SEQ + (HIST + t)) * DH;
  dst[i]      = (__bf16)(x1 * c - x2 * s);
  dst[i + 64] = (__bf16)(x2 * c + x1 * s);
}

// new V (f32 [n,HKV,DH]) -> bf16 V [b][h][1536+t][DH]
__global__ __launch_bounds__(256) void pack_v_kernel(
    const float* __restrict__ v, __bf16* __restrict__ vb) {
  int idx = blockIdx.x * blockDim.x + threadIdx.x;   // N*HKV*DH
  if (idx >= NTOK * HKV * DH) return;
  int d = idx & (DH - 1);
  int h = (idx >> 7) & (HKV - 1);
  int n = idx >> 10;
  int b = n / QO, t = n % QO;
  vb[(((size_t)b * HKV + h) * SEQ + (HIST + t)) * DH + d] =
      (__bf16)v[((size_t)n * HKV + h) * DH + d];
}

// gather paged f32 cache (pos < 1536) into dense bf16 K/V [b][h][pos][d]
__global__ __launch_bounds__(256) void gather_cache_kernel(
    const float* __restrict__ cache, const int* __restrict__ page_idx,
    const int* __restrict__ page_ptr, __bf16* __restrict__ kb,
    __bf16* __restrict__ vb) {
  int idx = blockIdx.x * blockDim.x + threadIdx.x;   // B*HKV*HIST*DH
  if (idx >= BB * HKV * HIST * DH) return;
  int d = idx & (DH - 1);
  int p = (idx >> 7) % HIST;
  int h = (idx / (DH * HIST)) & (HKV - 1);
  int b = idx / (DH * HIST * HKV);
  int page = page_idx[page_ptr[b] + (p >> 4)];
  int slot = p & (PAGE - 1);
  size_t base = ((((size_t)page * 2) * PAGE + slot) * HKV + h) * DH + d;
  size_t voff = ((((size_t)page * 2 + 1) * PAGE + slot) * HKV + h) * DH + d;
  size_t o = (((size_t)b * HKV + h) * SEQ + p) * DH + d;
  kb[o] = (__bf16)cache[base];
  vb[o] = (__bf16)cache[voff];
}

// ---------------------------------------------------------------------------
// Flash attention: one wave per (b, h, 16-query tile). Online softmax,
// S = Q K^T and O += P V both via v_wmma_f32_16x16x32_bf16.
// V chunk is staged with CDNA5 async global->LDS copies that overlap the
// QK^T WMMAs; consumed after s_wait_asynccnt 0.
// ---------------------------------------------------------------------------
__global__ __launch_bounds__(32) void attn_kernel(
    const __bf16* __restrict__ Q, const __bf16* __restrict__ Kc,
    const __bf16* __restrict__ Vc, float* __restrict__ O) {
  __shared__ __bf16 ldsP[16 * 32];      // P tile (16q x 32kv)
  __shared__ __bf16 ldsV[32 * 128];     // V chunk (32kv x 128d)

  const int qt = blockIdx.x;            // 0..31  (query tile)
  const int h  = blockIdx.y;            // 0..31
  const int b  = blockIdx.z;            // 0..3
  const int hk = h >> 2;                // GQA group of 4
  const int lane = threadIdx.x;
  const int half = lane >> 4, lm = lane & 15;

  // Q fragments: 4 x (16x32) covering D=128
  v16bf qa[4];
  {
    const __bf16* qrow =
        Q + (((size_t)b * HQ + h) * QO + qt * 16 + lm) * DH + half * 8;
#pragma unroll
    for (int f = 0; f < 4; ++f) {
      const __bf16* p = qrow + f * 32;
#pragma unroll
      for (int e = 0; e < 8; ++e) { qa[f][e] = p[e]; qa[f][e + 8] = p[e + 16]; }
    }
  }

  v8f acc[8] = {};
  float Mrow[8], Lrow[8];
#pragma unroll
  for (int j = 0; j < 8; ++j) { Mrow[j] = -1e30f; Lrow[j] = 0.0f; }

  const float scale = 0.08838834764831845f;   // 1/sqrt(128)
  const int q_pos_base = HIST + qt * 16;
  const int kv_end = q_pos_base + 16;         // causal: no kv beyond tile max
  const __bf16* Kb = Kc + ((size_t)b * HKV + hk) * SEQ * DH;
  const __bf16* Vb = Vc + ((size_t)b * HKV + hk) * SEQ * DH;
  const unsigned ldsV_base =
      (unsigned)(uintptr_t)(&ldsV[0]) + (unsigned)(lane * DH * 2);

  for (int kv0 = 0; kv0 < kv_end; kv0 += 32) {
    // issue async copy of V[kv0..kv0+31][0..127] -> LDS (overlaps QK^T below)
    {
      const char* vr = (const char*)(Vb + (size_t)(kv0 + lane) * DH);
#pragma unroll
      for (int u = 0; u < 16; ++u)
        async_load_lds_b128(ldsV_base + u * 16, vr + u * 16);
    }

    // scores: two 16x16 subtiles, each 4 WMMAs over D
    v8f sc[2];
#pragma unroll
    for (int s = 0; s < 2; ++s) {
      v8f st = {};
      const __bf16* krow =
          Kb + (size_t)(kv0 + s * 16 + lm) * DH + half * 8;
#pragma unroll
      for (int f = 0; f < 4; ++f) {
        v16bf kb;
        const __bf16* p = krow + f * 32;
#pragma unroll
        for (int e = 0; e < 8; ++e) { kb[e] = p[e]; kb[e + 8] = p[e + 16]; }
        st = __builtin_amdgcn_wmma_f32_16x16x32_bf16(
            false, qa[f], false, kb, (short)0, st, false, false);
      }
      sc[s] = st;
    }

    // mask + online softmax (row = j + half*8, col = lm across 16 lanes)
    float Mnew[8];
#pragma unroll
    for (int j = 0; j < 8; ++j) {
      int qp = q_pos_base + j + half * 8;
      float v0 = sc[0][j] * scale;
      float v1 = sc[1][j] * scale;
      if (kv0 + lm > qp)      v0 = -1e30f;
      if (kv0 + 16 + lm > qp) v1 = -1e30f;
      sc[0][j] = v0; sc[1][j] = v1;
      float mx = fmaxf(v0, v1);
#pragma unroll
      for (int d = 8; d >= 1; d >>= 1) mx = fmaxf(mx, __shfl_xor(mx, d, 32));
      Mnew[j] = fmaxf(Mrow[j], mx);
    }
#pragma unroll
    for (int j = 0; j < 8; ++j) {
      float alpha = __expf(Mrow[j] - Mnew[j]);
      float p0 = __expf(sc[0][j] - Mnew[j]);
      float p1 = __expf(sc[1][j] - Mnew[j]);
      float rs = p0 + p1;
#pragma unroll
      for (int d = 8; d >= 1; d >>= 1) rs += __shfl_xor(rs, d, 32);
      Lrow[j] = Lrow[j] * alpha + rs;
      Mrow[j] = Mnew[j];
#pragma unroll
      for (int t = 0; t < 8; ++t) acc[t][j] *= alpha;
      int m = j + half * 8;
      ldsP[m * 32 + lm]      = (__bf16)p0;
      ldsP[m * 32 + 16 + lm] = (__bf16)p1;
    }
    wait_asynccnt0();  // V chunk resident in LDS
    __syncthreads();   // single wave: orders ldsP stores vs loads (dscnt)

    // P A-fragment from LDS
    v16bf pa;
    {
      const __bf16* p = ldsP + lm * 32 + half * 8;
#pragma unroll
      for (int e = 0; e < 8; ++e) { pa[e] = p[e]; pa[e + 8] = p[e + 16]; }
    }
    // O(16x128) += P(16x32) * V(32x128): 8 WMMAs
#pragma unroll
    for (int t = 0; t < 8; ++t) {
      v16bf vb;
#pragma unroll
      for (int e = 0; e < 8; ++e) {
        vb[e] = ldsV[(half * 8 + e) * DH + t * 16 + lm];
        vb[e + 8] = ldsV[(16 + half * 8 + e) * DH + t * 16 + lm];
      }
      acc[t] = __builtin_amdgcn_wmma_f32_16x16x32_bf16(
          false, pa, false, vb, (short)0, acc[t], false, false);
    }
    __syncthreads();
  }

  // epilogue: O[n][h*DH + d] fp32
#pragma unroll
  for (int j = 0; j < 8; ++j) {
    int m = j + half * 8;
    float inv = 1.0f / Lrow[j];
    float* orow =
        O + (size_t)(b * QO + qt * 16 + m) * (HQ * DH) + h * DH;
#pragma unroll
    for (int t = 0; t < 8; ++t) orow[t * 16 + lm] = acc[t][j] * inv;
  }
}

// ---------------------------------------------------------------------------
// Launch: needs ~128MB of d_ws scratch.
// ---------------------------------------------------------------------------
extern "C" void kernel_launch(void* const* d_in, const int* in_sizes, int n_in,
                              void* d_out, int out_size, void* d_ws,
                              size_t ws_size, hipStream_t stream) {
  const float* hidden = (const float*)d_in[0];
  const int*   pos    = (const int*)d_in[1];
  const float* kvc    = (const float*)d_in[2];
  const int*   pidx   = (const int*)d_in[3];
  const int*   pptr   = (const int*)d_in[4];
  const float* Wq     = (const float*)d_in[7];
  const float* Wk     = (const float*)d_in[8];
  const float* Wv     = (const float*)d_in[9];
  const float* Wo     = (const float*)d_in[10];
  float* out = (float*)d_out;

  char* ws = (char*)d_ws;
  __bf16* hs_bf  = (__bf16*)(ws + 0);                      // 16 MB
  float*  q_f    = (float*)(ws + ((size_t)16 << 20));      // 32 MB
  float*  k_f    = (float*)(ws + ((size_t)48 << 20));      //  8 MB
  float*  v_f    = (float*)(ws + ((size_t)56 << 20));      //  8 MB
  __bf16* q_bf   = (__bf16*)(ws + ((size_t)64 << 20));     // 16 MB
  __bf16* K_bf   = (__bf16*)(ws + ((size_t)80 << 20));     // 16 MB
  __bf16* V_bf   = (__bf16*)(ws + ((size_t)96 << 20));     // 16 MB
  float*  attn_f = q_f;                                    // reuse q_f32
  __bf16* attn_bf = (__bf16*)(ws + ((size_t)112 << 20));   // 16 MB

  // 1) hidden f32 -> bf16
  cvt_f32_bf16_kernel<<<(NTOK * HID + 255) / 256, 256, 0, stream>>>(
      hidden, hs_bf, NTOK * HID);

  // 2) QKV projections (WMMA)
  gemm_bf16_kernel<<<dim3(NTOK / 16, (HQ * DH) / 512), 256, 0, stream>>>(
      hs_bf, Wq, q_f, NTOK, HID, HQ * DH);
  gemm_bf16_kernel<<<dim3(NTOK / 16, (HKV * DH) / 512), 256, 0, stream>>>(
      hs_bf, Wk, k_f, NTOK, HID, HKV * DH);
  gemm_bf16_kernel<<<dim3(NTOK / 16, (HKV * DH) / 512), 256, 0, stream>>>(
      hs_bf, Wv, v_f, NTOK, HID, HKV * DH);

  // 3) RoPE + pack new K/V; gather paged history into dense bf16 K/V
  rope_q_kernel<<<(NTOK * HQ * 64 + 255) / 256, 256, 0, stream>>>(
      q_f, pos, q_bf);
  rope_k_kernel<<<(NTOK * HKV * 64 + 255) / 256, 256, 0, stream>>>(
      k_f, pos, K_bf);
  pack_v_kernel<<<(NTOK * HKV * DH + 255) / 256, 256, 0, stream>>>(v_f, V_bf);
  gather_cache_kernel<<<(BB * HKV * HIST * DH + 255) / 256, 256, 0, stream>>>(
      kvc, pidx, pptr, K_bf, V_bf);

  // 4) flash attention (WMMA for QK^T and PV, async V staging)
  attn_kernel<<<dim3(QO / 16, HQ, BB), 32, 0, stream>>>(q_bf, K_bf, V_bf,
                                                        attn_f);

  // 5) output projection (WMMA)
  cvt_f32_bf16_kernel<<<(NTOK * HQ * DH + 255) / 256, 256, 0, stream>>>(
      attn_f, attn_bf, NTOK * HQ * DH);
  gemm_bf16_kernel<<<dim3(NTOK / 16, HID / 512), 256, 0, stream>>>(
      attn_bf, Wo, out, NTOK, HID, HID);
}
